// SuperpixelSubdomainClassifier_58875411693870
// MI455X (gfx1250) — compile-verified, matrix operand order checked by reading, not run on hardware
//
#include <hip/hip_runtime.h>
#include <hip/hip_bf16.h>

typedef __attribute__((ext_vector_type(16))) _Float16 v16h;
typedef __attribute__((ext_vector_type(8)))  float    v8f;

#define H_    512
#define W_    512
#define B_    4
#define CIN   3
#define COUT  256
#define HW_   (H_ * W_)            // 262144
#define NPIX  (B_ * HW_)           // 1048576
#define S_    1024
#define NSEG  2048
#define NS_TOT (B_ * NSEG)         // 8192

// VGPR-half -> K mapping for 16-bit WMMA A/B fragments (ISA 7.12.2):
// lanes 0-15:  V0..V3 hold K 0..7,  V4..V7 hold K 16..23
// lanes 16-31: V0..V3 hold K 8..15, V4..V7 hold K 24..31
__device__ __forceinline__ int kmap(int v, int h, int hi) {
  return ((v < 4) ? (2 * v) : (16 + 2 * (v - 4))) + h + (hi ? 8 : 0);
}

__device__ __forceinline__ void atomAddF(float* p, float v) {
  __hip_atomic_fetch_add(p, v, __ATOMIC_RELAXED, __HIP_MEMORY_SCOPE_AGENT);
}

__device__ __forceinline__ float block_sum256(float v, float* red, int tid) {
  red[tid] = v;
  __syncthreads();
  for (int s = 128; s > 0; s >>= 1) {
    if (tid < s) red[tid] += red[tid + s];
    __syncthreads();
  }
  float r = red[0];
  __syncthreads();
  return r;
}

// ---------------- prep: zero accumulators, convert weights to f16 ----------
__global__ void k_prep(float* wsf, _Float16* wh, _Float16* w1h, _Float16* w2h,
                       const float* __restrict__ cw, const float* __restrict__ w1,
                       const float* __restrict__ w2) {
  int t = blockIdx.x * 256 + threadIdx.x;
  if (t < 25152) wsf[t] = 0.0f;                       // diffsum + stats + centroid accum
  if (t < COUT * 32) {                                 // conv weights, K padded 27->32
    int oc = t >> 5, k = t & 31;
    wh[t] = (k < 27) ? (_Float16)cw[oc * 27 + k] : (_Float16)0.0f;
  }
  if (t < 128 * 256) w1h[t] = (_Float16)w1[t];         // w1 [128,256]
  if (t < 16 * 128) {                                  // w2 padded [8,128] -> [16,128]
    int m = t >> 7, d = t & 127;
    w2h[t] = (m < 8) ? (_Float16)w2[m * 128 + d] : (_Float16)0.0f;
  }
}

// ---- segmentation pass 1: block means, diff, global diff sum, x -> f16 ----
__global__ void k_seg_prep(const float* __restrict__ x, _Float16* __restrict__ xh,
                           float* __restrict__ diff, float* __restrict__ diffsum) {
  __shared__ float red[256];
  int blk = blockIdx.x, tid = threadIdx.x;
  int b = blk >> 10, t = blk & 1023, bh = t >> 5, bw = t & 31;
  int i = tid >> 4, j = tid & 15;
  int y = bh * 16 + i, xc = bw * 16 + j;
  int po = (y << 9) + xc;
  float v0 = x[((b * 3 + 0) << 18) + po];
  float v1 = x[((b * 3 + 1) << 18) + po];
  float v2 = x[((b * 3 + 2) << 18) + po];
  xh[((b * 3 + 0) << 18) + po] = (_Float16)v0;         // fused f16 conversion
  xh[((b * 3 + 1) << 18) + po] = (_Float16)v1;
  xh[((b * 3 + 2) << 18) + po] = (_Float16)v2;
  float m0 = block_sum256(v0, red, tid) * (1.0f / 256.0f);
  float m1 = block_sum256(v1, red, tid) * (1.0f / 256.0f);
  float m2 = block_sum256(v2, red, tid) * (1.0f / 256.0f);
  float d0 = v0 - m0, d1 = v1 - m1, d2 = v2 - m2;
  float dd = d0 * d0 + d1 * d1 + d2 * d2;
  diff[(b << 18) + po] = dd;
  float ds = block_sum256(dd, red, tid);
  if (tid == 0) atomAddF(diffsum, ds);
}

// ---------------- segmentation pass 2: labels + per-segment sums ----------
__global__ void k_seg_cent(const float* __restrict__ diff, const float* __restrict__ diffsum,
                           int* __restrict__ seg, float* __restrict__ cnt,
                           float* __restrict__ sy, float* __restrict__ sx) {
  __shared__ float red[256];
  int blk = blockIdx.x, tid = threadIdx.x;
  int b = blk >> 10, t = blk & 1023, bh = t >> 5, bw = t & 31;
  int i = tid >> 4, j = tid & 15;
  int y = bh * 16 + i, xc = bw * 16 + j;
  float thr = diffsum[0] * (2.0f / (float)NPIX);
  int p = (b << 18) + (y << 9) + xc;
  int fl = (diff[p] > thr) ? 1 : 0;
  int base = bh * 32 + bw;
  seg[p] = base + (fl ? S_ : 0);
  float co  = block_sum256((float)fl, red, tid);
  float syo = block_sum256(fl ? (float)y : 0.0f, red, tid);
  float sxo = block_sum256(fl ? (float)xc : 0.0f, red, tid);
  if (tid == 0) {
    // each 16x16 tile exclusively owns segments (base) and (base+S): direct stores
    int gb = b * NSEG + base, go = gb + S_;
    float syt = 256.0f * (float)(bh * 16) + 1920.0f;   // sum of y over tile
    float sxt = 256.0f * (float)(bw * 16) + 1920.0f;   // sum of x over tile
    cnt[gb] = 256.0f - co; sy[gb] = syt - syo; sx[gb] = sxt - sxo;
    cnt[go] = co;          sy[go] = syo;       sx[go] = sxo;
  }
}

// ---------------- centroids -> pixel index --------------------------------
__global__ void k_centroids(const float* __restrict__ cnt, const float* __restrict__ sy,
                            const float* __restrict__ sx, int* __restrict__ idx) {
  int t = blockIdx.x * 256 + threadIdx.x;
  if (t >= NS_TOT) return;
  float d = fmaxf(cnt[t], 1.0f);
  int cy = (int)(sy[t] / d); cy = min(max(cy, 0), H_ - 1);
  int cx = (int)(sx[t] / d); cx = min(max(cx, 0), W_ - 1);
  idx[t] = cy * W_ + cx;
}

// ---------------- WMMA implicit-GEMM conv -> BN statistics only ------------
// Never materializes f [4,256,512,512]; emits per-channel sum / sum-of-squares.
__global__ void __launch_bounds__(256) k_conv_stats(
    const _Float16* __restrict__ xh, const _Float16* __restrict__ wh,
    const float* __restrict__ cb, float* __restrict__ stat_s, float* __restrict__ stat_q) {
  const int lane = threadIdx.x & 31;
  const int wave = threadIdx.x >> 5;
  const int hi = lane >> 4;
  const int m  = lane & 15;
  const int ocg0 = wave, ocg1 = wave + 8;   // wave owns 32 disjoint channels

  // A fragments (resident) -----------------------------------------------
  v16h a0, a1;
#pragma unroll
  for (int i = 0; i < 16; ++i) {
    int k = kmap(i >> 1, i & 1, hi);
    a0[i] = wh[(ocg0 * 16 + m) * 32 + k];
    a1[i] = wh[(ocg1 * 16 + m) * 32 + k];
  }
  float bb0[8], bb1[8];
#pragma unroll
  for (int r = 0; r < 8; ++r) {
    bb0[r] = cb[ocg0 * 16 + r + hi * 8];
    bb1[r] = cb[ocg1 * 16 + r + hi * 8];
  }

  // per-lane loop-invariant patch offsets; K>=27 slots point at the tile
  // base pixel (finite value) -- safe since the matching A rows are zero.
  int off16[16];
#pragma unroll
  for (int i = 0; i < 16; ++i) {
    int k = kmap(i >> 1, i & 1, hi);
    int ci = k / 9, rr = k % 9;
    off16[i] = (k < 27) ? ((ci << 18) + (((rr / 3) - 1) << 9) + ((rr % 3) - 1)) : 0;
  }

  float s0[8] = {}, q0[8] = {}, s1[8] = {}, q1[8] = {};

  const int ntile = NPIX / 16;              // 65536 tiles of 16 row-contiguous pixels
  for (int t = blockIdx.x; t < ntile; t += gridDim.x) {
    int bimg = t >> 14;
    int rem  = t & 16383;
    int y    = rem >> 5;
    int x0   = (rem & 31) << 4;
    int xc   = x0 + m;                       // this lane's pixel column
    int pbase = ((bimg * 3) << 18) + (y << 9) + xc;
    v16h bf;
    if (y >= 1 && y <= 510 && x0 >= 16 && x0 <= 480) {
      // interior fast path: no bounds predication at all (wave-uniform branch)
#pragma unroll
      for (int i = 0; i < 16; ++i) bf[i] = xh[pbase + off16[i]];
    } else {
#pragma unroll
      for (int i = 0; i < 16; ++i) {
        int k = kmap(i >> 1, i & 1, hi);
        _Float16 val = (_Float16)0.0f;
        if (k < 27) {
          int ci = k / 9, rr = k % 9, ky = rr / 3, kx = rr % 3;
          int yy = y + ky - 1, xx = xc + kx - 1;
          if (yy >= 0 && yy < H_ && xx >= 0 && xx < W_)
            val = xh[((bimg * 3 + ci) << 18) + (yy << 9) + xx];
        }
        bf[i] = val;
      }
    }
    // prefetch next tile for this lane (global_prefetch_b8)
    int tn = t + gridDim.x;
    if (tn < ntile) {
      int bn = tn >> 14, rn = tn & 16383;
      __builtin_prefetch(xh + (((bn * 3) << 18) + ((rn >> 5) << 9) + ((rn & 31) << 4) + m), 0, 1);
    }
    v8f z = {};
    v8f d0 = __builtin_amdgcn_wmma_f32_16x16x32_f16(false, a0, false, bf, (short)0, z, false, false);
    v8f d1 = __builtin_amdgcn_wmma_f32_16x16x32_f16(false, a1, false, bf, (short)0, z, false, false);
#pragma unroll
    for (int r = 0; r < 8; ++r) {
      float v = d0[r] + bb0[r]; s0[r] += v; q0[r] += v * v;
      v = d1[r] + bb1[r];       s1[r] += v; q1[r] += v * v;
    }
  }
  // reduce across the 16 pixel lanes of each half-wave
#pragma unroll
  for (int r = 0; r < 8; ++r) {
#pragma unroll
    for (int msk = 1; msk < 16; msk <<= 1) {
      s0[r] += __shfl_xor(s0[r], msk, 16);
      q0[r] += __shfl_xor(q0[r], msk, 16);
      s1[r] += __shfl_xor(s1[r], msk, 16);
      q1[r] += __shfl_xor(q1[r], msk, 16);
    }
  }
  if (m == 0) {
#pragma unroll
    for (int r = 0; r < 8; ++r) {
      int oc0 = ocg0 * 16 + r + hi * 8;
      int oc1 = ocg1 * 16 + r + hi * 8;
      atomAddF(&stat_s[oc0], s0[r]); atomAddF(&stat_q[oc0], q0[r]);
      atomAddF(&stat_s[oc1], s1[r]); atomAddF(&stat_q[oc1], q1[r]);
    }
  }
}

// ---------------- fold BN into scale/shift --------------------------------
__global__ void k_bn(const float* __restrict__ stat_s, const float* __restrict__ stat_q,
                     const float* __restrict__ gamma, const float* __restrict__ beta,
                     float* __restrict__ scale, float* __restrict__ shift) {
  int c = threadIdx.x;
  float n = (float)NPIX;
  float mu  = stat_s[c] / n;
  float var = stat_q[c] / n - mu * mu;     // biased, matches reference
  float sc = gamma[c] * rsqrtf(var + 1e-5f);
  scale[c] = sc;
  shift[c] = beta[c] - mu * sc;
}

// ---------------- exact f32 conv+BN+relu at the 8192 centroid pixels ------
__global__ void k_cfeat(const float* __restrict__ x, const float* __restrict__ cw,
                        const float* __restrict__ cb, const float* __restrict__ scale,
                        const float* __restrict__ shift, const int* __restrict__ idx,
                        _Float16* __restrict__ g) {
  int s = blockIdx.x, oc = threadIdx.x;
  int bimg = s >> 11;
  int id = idx[s], cy = id >> 9, cx = id & 511;
  float acc = cb[oc];
#pragma unroll
  for (int k = 0; k < 27; ++k) {
    int ci = k / 9, rr = k % 9, ky = rr / 3, kx = rr % 3;
    int yy = cy + ky - 1, xx = cx + kx - 1;
    if (yy >= 0 && yy < H_ && xx >= 0 && xx < W_)
      acc += cw[oc * 27 + k] * x[((bimg * 3 + ci) << 18) + (yy << 9) + xx];
  }
  float f = fmaxf(acc * scale[oc] + shift[oc], 0.0f);
  g[s * 256 + oc] = (_Float16)f;
}

// ---------------- fused WMMA MLP + softmax: 16 segments per block ---------
__global__ void __launch_bounds__(256) k_mlp(
    const _Float16* __restrict__ g, const _Float16* __restrict__ w1h,
    const float* __restrict__ b1, const _Float16* __restrict__ w2h,
    const float* __restrict__ b2, float* __restrict__ probs) {
  __shared__ _Float16 hl[16 * 128];        // h tile [segment][d]
  const int lane = threadIdx.x & 31;
  const int wave = threadIdx.x >> 5;
  const int hi = lane >> 4;
  const int m  = lane & 15;
  const int st = blockIdx.x * 16;          // base segment of this tile

  // GEMM1: wave w computes h[:, 16w..16w+15] over K=256 (8 WMMA steps)
  v8f acc = {};
#pragma unroll
  for (int kc = 0; kc < 8; ++kc) {
    v16h af, bf;
#pragma unroll
    for (int i = 0; i < 16; ++i) {
      int k = kc * 32 + kmap(i >> 1, i & 1, hi);
      af[i] = w1h[(wave * 16 + m) * 256 + k];
      bf[i] = g[(st + m) * 256 + k];
    }
    acc = __builtin_amdgcn_wmma_f32_16x16x32_f16(false, af, false, bf, (short)0, acc, false, false);
  }
#pragma unroll
  for (int r = 0; r < 8; ++r) {
    int d = wave * 16 + r + hi * 8;
    float v = fmaxf(acc[r] + b1[d], 0.0f);
    hl[m * 128 + d] = (_Float16)v;
  }
  __syncthreads();

  // GEMM2 (wave 0): logits [16(pad of 8) x 16 segs] over K=128 (4 WMMA steps)
  if (wave == 0) {
    v8f acc2 = {};
#pragma unroll
    for (int kc = 0; kc < 4; ++kc) {
      v16h af, bf;
#pragma unroll
      for (int i = 0; i < 16; ++i) {
        int k = kc * 32 + kmap(i >> 1, i & 1, hi);
        af[i] = w2h[m * 128 + k];          // rows 8..15 are zero padding
        bf[i] = hl[m * 128 + k];
      }
      acc2 = __builtin_amdgcn_wmma_f32_16x16x32_f16(false, af, false, bf, (short)0, acc2, false, false);
    }
    if (lane < 16) {                        // lane n holds classes 0..7 of segment st+n
      float lg[8]; float mx = -1e30f;
#pragma unroll
      for (int r = 0; r < 8; ++r) { lg[r] = acc2[r] + b2[r]; mx = fmaxf(mx, lg[r]); }
      float den = 0.0f;
#pragma unroll
      for (int r = 0; r < 8; ++r) { lg[r] = __expf(lg[r] - mx); den += lg[r]; }
      float inv = 1.0f / den;
#pragma unroll
      for (int r = 0; r < 8; ++r) probs[(st + lane) * 8 + r] = lg[r] * inv;
    }
  }
}

// ---------------- per-pixel probability gather ----------------------------
__global__ void k_gather(const int* __restrict__ seg, const float* __restrict__ probs,
                         float* __restrict__ out) {
  int p = blockIdx.x * 256 + threadIdx.x;  // < NPIX
  int bimg = p >> 18;
  int pix = p & (HW_ - 1);
  const float* pr = probs + ((size_t)(bimg * NSEG + seg[p]) << 3);
  size_t ob = ((size_t)bimg << 21) + (size_t)pix;     // b*8*HW + pix
#pragma unroll
  for (int k = 0; k < 8; ++k) out[ob + ((size_t)k << 18)] = pr[k];
}

extern "C" void kernel_launch(void* const* d_in, const int* in_sizes, int n_in,
                              void* d_out, int out_size, void* d_ws, size_t ws_size,
                              hipStream_t stream) {
  (void)in_sizes; (void)n_in; (void)out_size; (void)ws_size;
  const float* x    = (const float*)d_in[0];
  const float* cw   = (const float*)d_in[1];
  const float* cb   = (const float*)d_in[2];
  const float* gmm  = (const float*)d_in[3];
  const float* beta = (const float*)d_in[4];
  const float* w1   = (const float*)d_in[5];
  const float* b1   = (const float*)d_in[6];
  const float* w2   = (const float*)d_in[7];
  const float* b2   = (const float*)d_in[8];
  float* out = (float*)d_out;

  char* ws = (char*)d_ws;
  float*    diffsum = (float*)(ws + 0);             // 1 f32 (padded to 256 B)
  float*    stat_s  = (float*)(ws + 256);           // 256 f32
  float*    stat_q  = (float*)(ws + 1280);          // 256 f32
  float*    cnt     = (float*)(ws + 2304);          // 8192 f32
  float*    sy      = (float*)(ws + 2304 + 32768);  // 8192 f32
  float*    sx      = (float*)(ws + 2304 + 65536);  // 8192 f32
  float*    scale   = (float*)(ws + 100608);        // 256 f32
  float*    shift   = (float*)(ws + 101632);        // 256 f32
  int*      idx     = (int*)  (ws + 102656);        // 8192 i32
  _Float16* wh      = (_Float16*)(ws + 135424);     // 256*32 f16
  _Float16* w1h     = (_Float16*)(ws + 151808);     // 128*256 f16
  _Float16* w2h     = (_Float16*)(ws + 217344);     // 16*128 f16
  float*    probs   = (float*)(ws + 221440);        // 8192*8 f32
  _Float16* xh      = (_Float16*)(ws + 483584);     // 3*NPIX f16
  float*    diff    = (float*)(ws + 6775040);       // NPIX f32
  int*      seg     = (int*)  (ws + 10969344);      // NPIX i32
  _Float16* g       = (_Float16*)(ws + 15163648);   // 8192*256 f16  (end: 19357952 B)

  k_prep      <<<128, 256, 0, stream>>>((float*)ws, wh, w1h, w2h, cw, w1, w2);
  k_seg_prep  <<<4096, 256, 0, stream>>>(x, xh, diff, diffsum);
  k_seg_cent  <<<4096, 256, 0, stream>>>(diff, diffsum, seg, cnt, sy, sx);
  k_centroids <<<32, 256, 0, stream>>>(cnt, sy, sx, idx);
  k_conv_stats<<<512, 256, 0, stream>>>(xh, wh, cb, stat_s, stat_q);
  k_bn        <<<1, 256, 0, stream>>>(stat_s, stat_q, gmm, beta, scale, shift);
  k_cfeat     <<<8192, 256, 0, stream>>>(x, cw, cb, scale, shift, idx, g);
  k_mlp       <<<512, 256, 0, stream>>>(g, w1h, b1, w2h, b2, probs);
  k_gather    <<<4096, 256, 0, stream>>>(seg, probs, out);
}
